// PhysicsAugEGATBlock_86088324481791
// MI455X (gfx1250) — matrix operand order repaired and checked by reading, not compiled, not run on hardware
//
#include <hip/hip_runtime.h>
#include <hip/hip_bf16.h>

typedef __attribute__((ext_vector_type(16))) _Float16 v16h;
typedef __attribute__((ext_vector_type(8)))  _Float16 v8h;
typedef __attribute__((ext_vector_type(2)))  _Float16 v2h;
typedef __attribute__((ext_vector_type(8)))  float    v8f;

#define NN 50000
#define EE 800000
#define DD 64
#define HH 4
#define HD 256           // H*D
#define NEG_SLOPE 0.2f

// ---- monotonic float <-> uint key for atomicMax-based segment max ----
__device__ __forceinline__ unsigned fkey(float f) {
  unsigned b = __float_as_uint(f);
  return (b & 0x80000000u) ? ~b : (b | 0x80000000u);
}
__device__ __forceinline__ float fdecode(unsigned k) {
  unsigned b = (k & 0x80000000u) ? (k & 0x7FFFFFFFu) : ~k;
  return __uint_as_float(b);
}

// ---- WMMA fragment builders per CDNA5 16-bit VGPR layouts (ISA 7.12.2) ----
// A matrix 16x32 f16: lanes 0-15 row M=lane hold K {0..7,16..23}; lanes 16-31 K {8..15,24..31}
// Per-lane data = two contiguous 32B runs (f32) / 16B runs (f16) -> vector loads.
__device__ __forceinline__ v16h a_frag_f32(const float* base, int ld, int kofs, int lane) {
  int r  = lane & 15;
  int kb = (lane & 16) ? 8 : 0;
  const float* row = base + r * ld + kofs + kb;
  v8f lo = *(const v8f*)(row);
  v8f hi = *(const v8f*)(row + 16);
  v16h a;
#pragma unroll
  for (int i = 0; i < 8; ++i) { a[i] = (_Float16)lo[i]; a[8 + i] = (_Float16)hi[i]; }
  return a;
}
__device__ __forceinline__ v16h a_frag_h16(const _Float16* base, int ld, int kofs, int lane) {
  int r  = lane & 15;
  int kb = (lane & 16) ? 8 : 0;
  const _Float16* row = base + r * ld + kofs + kb;
  v8h lo = *(const v8h*)(row);
  v8h hi = *(const v8h*)(row + 16);
  v16h a;
#pragma unroll
  for (int i = 0; i < 8; ++i) { a[i] = lo[i]; a[8 + i] = hi[i]; }
  return a;
}
// B fragment from pre-packed f16 weights: one contiguous 32B load per lane.
__device__ __forceinline__ v16h b_frag_pk(const _Float16* Wp, int ct, int nkc, int kc, int lane) {
  return *(const v16h*)(Wp + ((((size_t)ct * nkc + kc) << 5) + lane) * 16);
}
__device__ __forceinline__ v8f wmma32(v16h a, v16h b, v8f c) {
  return __builtin_amdgcn_wmma_f32_16x16x32_f16(false, a, false, b, (short)0, c, false, false);
}

// ============== Kernel 0: pack fp32 weight (K x ncols) into B-fragment order
// out[((ct*nkc + kc)*32 + lane)*16 + i] = (f16) W[k*ld + n]
//   n = ct*16 + (lane&15), k = kc*32 + ((lane&16)?8:0) + (i&7) + ((i&8)?16:0).  16384 elems/matrix.
__global__ __launch_bounds__(256) void egat_pack_b(const float* __restrict__ W, int ld,
                                                   int nkc, _Float16* __restrict__ out) {
  int idx  = blockIdx.x * 256 + threadIdx.x;
  int i    = idx & 15;
  int lane = (idx >> 4) & 31;
  int grp  = idx >> 9;                 // ct*nkc + kc  (0..31)
  int kc   = grp % nkc;
  int ct   = grp / nkc;
  int n = ct * 16 + (lane & 15);
  int k = kc * 32 + ((lane & 16) ? 8 : 0) + (i & 7) + ((i & 8) ? 16 : 0);
  out[idx] = (_Float16)W[(size_t)k * ld + n];
}

// ======================= Kernel 1: node GEMMs ==============================
// P_ni = hE@W_ni ; P_nj = hE@W_nj ; Pproj = hE@W_node + b_node   (all N x 256)
__global__ __launch_bounds__(128) void egat_node_gemm(
    const float* __restrict__ hE, const _Float16* __restrict__ Wp_ni,
    const _Float16* __restrict__ Wp_nj, const _Float16* __restrict__ Wp_nd,
    const float* __restrict__ b_node,
    float* __restrict__ P_ni, float* __restrict__ P_nj, float* __restrict__ Pproj)
{
  int tile = blockIdx.x;            // 16-node tile
  int wave = threadIdx.x >> 5;
  int lane = threadIdx.x & 31;
  const float* Abase = hE + (size_t)tile * 16 * DD;
  v16h a0 = a_frag_f32(Abase, DD, 0, lane);
  v16h a1 = a_frag_f32(Abase, DD, 32, lane);
  int mrow = (lane & 16) ? 8 : 0;
  int nloc = lane & 15;
#pragma unroll
  for (int c = 0; c < 4; ++c) {
    int ct = wave * 4 + c;
    int col0 = ct * 16;
    v8f acc_i = {}, acc_j = {}, acc_p = {};
    acc_i = wmma32(a0, b_frag_pk(Wp_ni, ct, 2, 0, lane), acc_i);
    acc_i = wmma32(a1, b_frag_pk(Wp_ni, ct, 2, 1, lane), acc_i);
    acc_j = wmma32(a0, b_frag_pk(Wp_nj, ct, 2, 0, lane), acc_j);
    acc_j = wmma32(a1, b_frag_pk(Wp_nj, ct, 2, 1, lane), acc_j);
    acc_p = wmma32(a0, b_frag_pk(Wp_nd, ct, 2, 0, lane), acc_p);
    acc_p = wmma32(a1, b_frag_pk(Wp_nd, ct, 2, 1, lane), acc_p);
    float bias = b_node[col0 + nloc];
#pragma unroll
    for (int p = 0; p < 8; ++p) {
      size_t row = (size_t)tile * 16 + p + mrow;
      size_t idx = row * HD + col0 + nloc;
      P_ni[idx]  = acc_i[p];
      P_nj[idx]  = acc_j[p];
      Pproj[idx] = acc_p[p] + bias;
    }
  }
}

// ======================= Kernel 2: fused edge kernel =======================
// psum = P_ni[src]+P_nj[dst] staged in LDS (vector loads, rows read once);
// f = leaky(hH@W_fij + psum) kept in LDS (f16);
// e = f . attn (per head) -> e_buf + atomicMax emax key;
// out_edge = f@Wp_edge + bp_edge + hH  (residual).  One 16-edge tile / block.
__global__ __launch_bounds__(128) void egat_edge(
    const float* __restrict__ hH, const int* __restrict__ src, const int* __restrict__ dst,
    const _Float16* __restrict__ Wp_fij, const float* __restrict__ attn,
    const _Float16* __restrict__ Wp_pe, const float* __restrict__ bp_edge,
    const float* __restrict__ P_ni, const float* __restrict__ P_nj,
    float* __restrict__ e_buf, unsigned* __restrict__ emaxk,
    float* __restrict__ out_edge)
{
  __shared__ __align__(32) float    hh[16 * DD];     // 4 KB edge-feature tile
  __shared__ __align__(32) float    psum[16 * HD];   // 16 KB gathered row sums
  __shared__ __align__(32) _Float16 ftile[16 * HD];  // 8 KB f tile (f16)
  __shared__ int s_src[16], s_dst[16];

  int tile = blockIdx.x;
  size_t e0 = (size_t)tile * 16;
  int tid  = threadIdx.x;
  int wave = tid >> 5;                 // wave == head
  int lane = tid & 31;

  *(v8f*)(hh + tid * 8) = *(const v8f*)(hH + e0 * DD + tid * 8);
  if (tid < 16)      s_src[tid]      = src[e0 + tid];
  else if (tid < 32) s_dst[tid - 16] = dst[e0 + tid - 16];
  __syncthreads();

  // cooperative stage: psum[m][*] = P_ni[src[m]][*] + P_nj[dst[m]][*]
  // 512 chunks of 8 floats; 4 per thread, all b128-pair vector loads.
#pragma unroll
  for (int it = 0; it < 4; ++it) {
    int chunk = tid + it * 128;
    int m   = chunk >> 5;
    int off = (chunk & 31) * 8;
    v8f va = *(const v8f*)(P_ni + (size_t)s_src[m] * HD + off);
    v8f vb = *(const v8f*)(P_nj + (size_t)s_dst[m] * HD + off);
    *(v8f*)(psum + m * HD + off) = va + vb;
  }

  v16h a0 = a_frag_f32(hh, DD, 0, lane);   // overlaps with staging loads
  v16h a1 = a_frag_f32(hh, DD, 32, lane);
  int mrow = (lane & 16) ? 8 : 0;
  int nloc = lane & 15;
  float eacc[8];
#pragma unroll
  for (int p = 0; p < 8; ++p) eacc[p] = 0.f;
  __syncthreads();                         // psum ready

#pragma unroll
  for (int c = 0; c < 4; ++c) {
    int ct = wave * 4 + c;             // this wave covers its head's 64 columns
    int col0 = ct * 16;
    v8f acc = {};
    acc = wmma32(a0, b_frag_pk(Wp_fij, ct, 2, 0, lane), acc);
    acc = wmma32(a1, b_frag_pk(Wp_fij, ct, 2, 1, lane), acc);
    int col = col0 + nloc;
    float av = attn[wave * DD + (col & (DD - 1))];
#pragma unroll
    for (int p = 0; p < 8; ++p) {
      int m = p + mrow;
      float g = acc[p] + psum[m * HD + col];
      float f = fmaxf(g, 0.f) + NEG_SLOPE * fminf(g, 0.f);   // branchless leaky
      eacc[p] += f * av;
      ftile[m * HD + col] = (_Float16)f;
    }
  }
  // reduce logits across the 16-lane half-groups (xor masks < 16 stay in half)
#pragma unroll
  for (int off = 8; off >= 1; off >>= 1) {
#pragma unroll
    for (int p = 0; p < 8; ++p) eacc[p] += __shfl_xor(eacc[p], off, 32);
  }
  if (nloc == 0) {
#pragma unroll
    for (int p = 0; p < 8; ++p) {
      int m = p + mrow;
      float ev = eacc[p];
      e_buf[(e0 + m) * HH + wave] = ev;
      atomicMax(&emaxk[(size_t)s_dst[m] * HH + wave], fkey(ev));
    }
  }
  __syncthreads();   // ftile complete; EXEC reconverged before WMMA

  // hH_new = f @ Wp_edge + bp_edge + hH  (K=256, each wave one 16-col tile)
  v8f acc2 = {};
#pragma unroll
  for (int kc = 0; kc < 8; ++kc) {
    v16h af = a_frag_h16(ftile, HD, kc * 32, lane);
    v16h bf = b_frag_pk(Wp_pe, wave, 8, kc, lane);
    acc2 = wmma32(af, bf, acc2);
  }
  int col = wave * 16 + nloc;
  float bias = bp_edge[col];
#pragma unroll
  for (int p = 0; p < 8; ++p) {
    int m = p + mrow;
    out_edge[(e0 + m) * DD + col] = acc2[p] + bias + hh[m * DD + col];
  }
}

// =============== Kernel 3: exp + scatter (unnormalized softmax) ============
// w = exp(e - emax[dst]); ssum[dst,h] += w; agg[dst, h*64+d] += h_proj[src]*w
__global__ __launch_bounds__(256) void egat_scatter(
    const int* __restrict__ src, const int* __restrict__ dst,
    const float* __restrict__ e_buf, const unsigned* __restrict__ emaxk,
    const float* __restrict__ Pproj,
    float* __restrict__ ssum, float* __restrict__ agg)
{
  size_t g = (size_t)blockIdx.x * blockDim.x + threadIdx.x;   // E*32 threads
  size_t edge = g >> 5;
  int j = (int)(g & 31);                // 32 threads/edge, 8 cols each
  if (edge >= (size_t)EE) return;
  int h = j >> 3;
  int s = src[edge], d = dst[edge];
  float ev = e_buf[edge * HH + h];
  float mx = fdecode(emaxk[(size_t)d * HH + h]);
  float w = __expf(ev - mx);
  if ((j & 7) == 0) atomicAdd(&ssum[(size_t)d * HH + h], w);
  int c0 = j * 8;
  v8f pv = *(const v8f*)(Pproj + (size_t)s * HD + c0);        // one b128x2 load
  float* ag = agg + (size_t)d * HD + c0;
#pragma unroll
  for (int i = 0; i < 8; ++i) atomicAdd(&ag[i], pv[i] * w);
}

// =================== Kernel 4: normalize + node projection =================
// hE_out = (agg/ssum) @ Wp_node + bp_node + hE
__global__ __launch_bounds__(128) void egat_node_out(
    const float* __restrict__ hE, const float* __restrict__ agg,
    const float* __restrict__ ssum, const _Float16* __restrict__ Wp_pn,
    const float* __restrict__ bp_node, float* __restrict__ out_node)
{
  __shared__ __align__(32) _Float16 ntile[16 * HD];  // 8 KB normalized tile
  int tile = blockIdx.x;
  size_t n0 = (size_t)tile * 16;
  int tid = threadIdx.x;
  for (int i = tid * 2; i < 16 * HD; i += 256) {     // packed v2h writes
    int m = i >> 8, n = i & 255;
    size_t row = n0 + m;
    float s0 = ssum[row * HH + (n >> 6)];
    float v0 = agg[row * HD + n];
    float v1 = agg[row * HD + n + 1];
    float inv = (s0 > 0.f) ? (1.f / s0) : 0.f;       // (n, n+1 share head)
    v2h hv; hv[0] = (_Float16)(v0 * inv); hv[1] = (_Float16)(v1 * inv);
    *(v2h*)(ntile + i) = hv;
  }
  __syncthreads();
  int wave = tid >> 5, lane = tid & 31;
  int mrow = (lane & 16) ? 8 : 0, nloc = lane & 15;
  v8f acc = {};
#pragma unroll
  for (int kc = 0; kc < 8; ++kc) {
    v16h a = a_frag_h16(ntile, HD, kc * 32, lane);
    v16h b = b_frag_pk(Wp_pn, wave, 8, kc, lane);
    acc = wmma32(a, b, acc);
  }
  int col = wave * 16 + nloc;
  float bias = bp_node[col];
#pragma unroll
  for (int p = 0; p < 8; ++p) {
    size_t row = n0 + p + mrow;
    out_node[row * DD + col] = acc[p] + bias + hE[row * DD + col];
  }
}

// ============================== launcher ===================================
extern "C" void kernel_launch(void* const* d_in, const int* in_sizes, int n_in,
                              void* d_out, int out_size, void* d_ws, size_t ws_size,
                              hipStream_t stream) {
  (void)in_sizes; (void)n_in; (void)out_size; (void)ws_size;
  const float* hE      = (const float*)d_in[0];
  const float* hH      = (const float*)d_in[1];
  const int*   src     = (const int*)d_in[2];
  const int*   dst     = (const int*)d_in[3];
  const float* W_node  = (const float*)d_in[4];
  const float* b_node  = (const float*)d_in[5];
  const float* W_ni    = (const float*)d_in[6];
  const float* W_nj    = (const float*)d_in[7];
  const float* W_fij   = (const float*)d_in[8];
  const float* attn    = (const float*)d_in[9];
  const float* Wp_node = (const float*)d_in[10];
  const float* bp_node = (const float*)d_in[11];
  const float* Wp_edge = (const float*)d_in[12];
  const float* bp_edge = (const float*)d_in[13];

  float* out_node = (float*)d_out;                 // N*64
  float* out_edge = out_node + (size_t)NN * DD;    // then E*64

  float* ws    = (float*)d_ws;
  float* P_ni  = ws;  ws += (size_t)NN * HD;
  float* P_nj  = ws;  ws += (size_t)NN * HD;
  float* Pproj = ws;  ws += (size_t)NN * HD;
  float* e_buf = ws;  ws += (size_t)EE * HH;
  float* zbase = ws;                               // contiguous zero region:
  float* agg   = ws;  ws += (size_t)NN * HD;       //   agg
  float* ssum  = ws;  ws += (size_t)NN * HH;       //   ssum
  unsigned* emaxk = (unsigned*)ws; ws += (size_t)NN * HH; // emax keys (0 < key(x))

  // packed f16 weights (fragment order), 16384 halves each
  _Float16* pk = (_Float16*)ws;
  _Float16* pk_ni  = pk;  pk += 16384;
  _Float16* pk_nj  = pk;  pk += 16384;
  _Float16* pk_nd  = pk;  pk += 16384;
  _Float16* pk_fij = pk;  pk += 16384;
  _Float16* pk_pe  = pk;  pk += 16384;
  _Float16* pk_pn  = pk;  pk += 16384;

  size_t zbytes = ((size_t)NN * HD + 2 * (size_t)NN * HH) * sizeof(float);
  hipMemsetAsync(zbase, 0, zbytes, stream);

  // pack weights into WMMA B-fragment order (f16)
  egat_pack_b<<<64, 256, 0, stream>>>(W_ni,   HD, 2, pk_ni);
  egat_pack_b<<<64, 256, 0, stream>>>(W_nj,   HD, 2, pk_nj);
  egat_pack_b<<<64, 256, 0, stream>>>(W_node, HD, 2, pk_nd);
  egat_pack_b<<<64, 256, 0, stream>>>(W_fij,  HD, 2, pk_fij);
  egat_pack_b<<<64, 256, 0, stream>>>(Wp_edge, DD, 8, pk_pe);
  egat_pack_b<<<64, 256, 0, stream>>>(Wp_node, DD, 8, pk_pn);

  egat_node_gemm<<<NN / 16, 128, 0, stream>>>(hE, pk_ni, pk_nj, pk_nd, b_node,
                                              P_ni, P_nj, Pproj);
  egat_edge<<<EE / 16, 128, 0, stream>>>(hH, src, dst, pk_fij, attn, pk_pe, bp_edge,
                                         P_ni, P_nj, e_buf, emaxk, out_edge);
  egat_scatter<<<(EE * 32) / 256, 256, 0, stream>>>(src, dst, e_buf, emaxk, Pproj,
                                                    ssum, agg);
  egat_node_out<<<NN / 16, 128, 0, stream>>>(hE, agg, ssum, pk_pn, bp_node, out_node);
}
